// Decoder_70188355551677
// MI455X (gfx1250) — matrix operand order repaired and checked by reading, not compiled, not run on hardware
//
#include <hip/hip_runtime.h>
#include <math.h>

typedef __bf16 bf16_t;
typedef __attribute__((ext_vector_type(16))) __bf16 v16bf;
typedef __attribute__((ext_vector_type(8)))  float  v8f;
typedef __attribute__((ext_vector_type(4)))  float  f32x4;

#define SEQ_LEN 128
#define BATCH   256
#define HDIM    512

// ======================= weight prep =======================
__global__ void k_cvt_bf16(const float* __restrict__ src, bf16_t* __restrict__ dst, int n) {
  int i = blockIdx.x * blockDim.x + threadIdx.x;
  if (i < n) dst[i] = (bf16_t)src[i];
}

// Wcat[j][0:512]=Wi[j], Wcat[j][512:1024]=Wh[j]  (2048 x 1024)
__global__ void k_prep_wcat(const float* __restrict__ Wi, const float* __restrict__ Wh,
                            bf16_t* __restrict__ Wcat) {
  int i = blockIdx.x * blockDim.x + threadIdx.x;   // exactly 2048*1024 threads
  int j = i >> 10, k = i & 1023;
  float v = (k < 512) ? Wi[j * 512 + k] : Wh[j * 512 + (k - 512)];
  Wcat[i] = (bf16_t)v;
}

__global__ void k_prep_biascat(const float* __restrict__ bi, const float* __restrict__ bh,
                               float* __restrict__ bc) {
  int i = blockIdx.x * blockDim.x + threadIdx.x;
  if (i < 2048) bc[i] = bi[i] + bh[i];
}

// ======================= WMMA GEMM =======================
// A fragment (16x32 bf16): lane -> row, lanes 0-15 K{0..15}, lanes 16-31 K{16..31}.
// p must already be offset by (row*lda + half*16); 16 contiguous k.
__device__ __forceinline__ v16bf make_afrag(const float* __restrict__ p) {
  f32x4 a0 = *(const f32x4*)p;
  f32x4 a1 = *(const f32x4*)(p + 4);
  f32x4 a2 = *(const f32x4*)(p + 8);
  f32x4 a3 = *(const f32x4*)(p + 12);
  v16bf af;
  af[0]=(bf16_t)a0.x;  af[1]=(bf16_t)a0.y;  af[2]=(bf16_t)a0.z;  af[3]=(bf16_t)a0.w;
  af[4]=(bf16_t)a1.x;  af[5]=(bf16_t)a1.y;  af[6]=(bf16_t)a1.z;  af[7]=(bf16_t)a1.w;
  af[8]=(bf16_t)a2.x;  af[9]=(bf16_t)a2.y;  af[10]=(bf16_t)a2.z; af[11]=(bf16_t)a2.w;
  af[12]=(bf16_t)a3.x; af[13]=(bf16_t)a3.y; af[14]=(bf16_t)a3.z; af[15]=(bf16_t)a3.w;
  return af;
}

// C[m, n] = sum_k A[m,k]*W[n,k] + bias[n], with A = [A0 (K0 cols) | A1 (K1 cols)].
// W bf16 [N][K0+K1] row-major. Block tile 64x128, 8 waves, each wave: 1 M-tile x 4 N-tiles.
// Requires M%64==0, N%128==0, K0%32==0, K1%32==0.
// permuteBL: output row = (m%BATCH)*SEQ_LEN + m/BATCH  (e_g/e_p stored [b][l][h]).
__global__ __launch_bounds__(256) void gemm_bf16_wmma(
    const float* __restrict__ A0, const float* __restrict__ A1,
    int lda, int K0, int K1,
    const bf16_t* __restrict__ W, const float* __restrict__ bias,
    float* __restrict__ C, int ldc, int permuteBL)
{
  const int lane = threadIdx.x & 31;
  const int wv   = threadIdx.x >> 5;          // 8 waves
  const int half = lane >> 4;                 // 0|1
  const int l15  = lane & 15;
  const int K    = K0 + K1;
  const int mTile = blockIdx.y * 64 + (wv & 3) * 16;
  const int nBase = blockIdx.x * 128 + (wv >> 2) * 64;
  const int mrow  = mTile + l15;

  // B fragment rows: lane -> n, 16 contiguous k per half-wave (same k order as A)
  const bf16_t* Wr0 = W + (size_t)(nBase +  0 + l15) * K + half * 16;
  const bf16_t* Wr1 = W + (size_t)(nBase + 16 + l15) * K + half * 16;
  const bf16_t* Wr2 = W + (size_t)(nBase + 32 + l15) * K + half * 16;
  const bf16_t* Wr3 = W + (size_t)(nBase + 48 + l15) * K + half * 16;

  v8f acc0 = {0.f,0.f,0.f,0.f,0.f,0.f,0.f,0.f};
  v8f acc1 = acc0, acc2 = acc0, acc3 = acc0;

  // ---- phase 0: A0, k in [0, K0) ----
  {
    const float* pA = A0 + (size_t)mrow * lda + half * 16;
    for (int k0 = 0; k0 < K0; k0 += 32) {
      v16bf af = make_afrag(pA + k0);
      v16bf b0 = *(const v16bf*)(Wr0 + k0);
      v16bf b1 = *(const v16bf*)(Wr1 + k0);
      v16bf b2 = *(const v16bf*)(Wr2 + k0);
      v16bf b3 = *(const v16bf*)(Wr3 + k0);
      if (k0 + 32 < K) __builtin_prefetch(Wr0 + k0 + 32, 0, 1);
      acc0 = __builtin_amdgcn_wmma_f32_16x16x32_bf16(false, af, false, b0, (short)0, acc0, false, false);
      acc1 = __builtin_amdgcn_wmma_f32_16x16x32_bf16(false, af, false, b1, (short)0, acc1, false, false);
      acc2 = __builtin_amdgcn_wmma_f32_16x16x32_bf16(false, af, false, b2, (short)0, acc2, false, false);
      acc3 = __builtin_amdgcn_wmma_f32_16x16x32_bf16(false, af, false, b3, (short)0, acc3, false, false);
    }
  }
  // ---- phase 1: A1, k in [0, K1), W columns offset by K0 ----
  {
    const float* pA = A1 + (size_t)mrow * lda + half * 16;
    for (int k0 = 0; k0 < K1; k0 += 32) {
      v16bf af = make_afrag(pA + k0);
      v16bf b0 = *(const v16bf*)(Wr0 + K0 + k0);
      v16bf b1 = *(const v16bf*)(Wr1 + K0 + k0);
      v16bf b2 = *(const v16bf*)(Wr2 + K0 + k0);
      v16bf b3 = *(const v16bf*)(Wr3 + K0 + k0);
      if (k0 + 32 < K1) __builtin_prefetch(Wr0 + K0 + k0 + 32, 0, 1);
      acc0 = __builtin_amdgcn_wmma_f32_16x16x32_bf16(false, af, false, b0, (short)0, acc0, false, false);
      acc1 = __builtin_amdgcn_wmma_f32_16x16x32_bf16(false, af, false, b1, (short)0, acc1, false, false);
      acc2 = __builtin_amdgcn_wmma_f32_16x16x32_bf16(false, af, false, b2, (short)0, acc2, false, false);
      acc3 = __builtin_amdgcn_wmma_f32_16x16x32_bf16(false, af, false, b3, (short)0, acc3, false, false);
    }
  }

  // C/D layout: lane<16 -> N=lane, M=mTile+r ; lane>=16 -> N=lane-16, M=mTile+8+r
  for (int r = 0; r < 8; ++r) {
    int m   = mTile + half * 8 + r;
    int row = permuteBL ? ((m & (BATCH - 1)) * SEQ_LEN + (m >> 8)) : m;
    float* crow = C + (size_t)row * ldc;
    int n = nBase + l15;
    crow[n     ] = acc0[r] + bias[n     ];
    crow[n + 16] = acc1[r] + bias[n + 16];
    crow[n + 32] = acc2[r] + bias[n + 32];
    crow[n + 48] = acc3[r] + bias[n + 48];
  }
}

// ======================= LSTM pointwise =======================
__device__ __forceinline__ float sigmoidf_(float x) { return 1.0f / (1.0f + expf(-x)); }

__global__ void k_lstm_pw(const float* __restrict__ gates,
                          float* __restrict__ hx, float* __restrict__ cx) {
  int i = blockIdx.x * blockDim.x + threadIdx.x;  // BATCH*HDIM = 131072 threads
  int b = i >> 9, h = i & 511;
  const float* g = gates + (size_t)b * 2048;
  float ig = sigmoidf_(g[h]);
  float fg = sigmoidf_(g[512 + h]);
  float gg = tanhf(g[1024 + h]);
  float og = sigmoidf_(g[1536 + h]);
  float c  = fg * cx[i] + ig * gg;
  cx[i] = c;
  hx[i] = og * tanhf(c);
}

// ======================= attention =======================
// e stored [b][l][h] (h contiguous).
__global__ __launch_bounds__(128) void k_glimpse(
    const float* __restrict__ e, const float* __restrict__ q,
    const float* __restrict__ v, const unsigned* __restrict__ mask,
    float* __restrict__ gl)
{
  __shared__ float sQ[HDIM];
  __shared__ float sV[HDIM];
  __shared__ float sU[SEQ_LEN];
  __shared__ float sR[4];
  const int b = blockIdx.x;
  const int tid = threadIdx.x, lane = tid & 31, wv = tid >> 5;
  const float* eb = e + (size_t)b * SEQ_LEN * HDIM;
  const float* qb = q + (size_t)b * HDIM;

  for (int h = tid; h < HDIM; h += 128) { sQ[h] = qb[h]; sV[h] = v[h]; }
  __syncthreads();

  // u[l] = sum_h v[h]*tanh(q[h] + e[b,l,h]); one wave per l, lanes over h
  for (int l = wv; l < SEQ_LEN; l += 4) {
    const float* el = eb + (size_t)l * HDIM;
    float p = 0.f;
    for (int h = lane; h < HDIM; h += 32)
      p += sV[h] * tanhf(sQ[h] + el[h]);
    for (int o = 16; o; o >>= 1) p += __shfl_xor(p, o, 32);
    if (lane == 0) sU[l] = mask[b * SEQ_LEN + l] ? -1e30f : p;
  }
  __syncthreads();

  // softmax over 128 (one l per thread)
  float u = sU[tid];
  float m = u;
  for (int o = 16; o; o >>= 1) m = fmaxf(m, __shfl_xor(m, o, 32));
  if (lane == 0) sR[wv] = m;
  __syncthreads();
  m = fmaxf(fmaxf(sR[0], sR[1]), fmaxf(sR[2], sR[3]));
  float ex = expf(u - m);
  float s = ex;
  for (int o = 16; o; o >>= 1) s += __shfl_xor(s, o, 32);
  __syncthreads();
  if (lane == 0) sR[wv] = s;
  __syncthreads();
  s = sR[0] + sR[1] + sR[2] + sR[3];
  sU[tid] = ex / s;
  __syncthreads();

  // gl[b,h] = sum_l a[l] * e[b,l,h]
  for (int h = tid; h < HDIM; h += 128) {
    float acc = 0.f;
    for (int l = 0; l < SEQ_LEN; ++l)
      acc += sU[l] * eb[(size_t)l * HDIM + h];
    gl[(size_t)b * HDIM + h] = acc;
  }
}

__device__ __forceinline__ float gumbel_noise(unsigned t, unsigned b, unsigned l) {
  unsigned x = t * 0x9E3779B9u ^ (b * 0x85EBCA6Bu + 0x165667B1u)
                              ^ (l * 0xC2B2AE35u + 0x27D4EB2Fu);
  x ^= x >> 16; x *= 0x7FEB352Du; x ^= x >> 15; x *= 0x846CA68Bu; x ^= x >> 16;
  float u = (float)(x >> 8) * (1.0f / 16777216.0f);
  u = fmaxf(u, 1e-10f);
  return -logf(-logf(u));
}

__global__ __launch_bounds__(128) void k_pointer(
    const float* __restrict__ e, const float* __restrict__ q,
    const float* __restrict__ v, unsigned* __restrict__ mask,
    int* __restrict__ prev, float* __restrict__ probs_out,
    float* __restrict__ sel_out, int t)
{
  __shared__ float sQ[HDIM];
  __shared__ float sV[HDIM];
  __shared__ float sU[SEQ_LEN];
  __shared__ float sR[4];
  __shared__ int   sI[4];
  const int b = blockIdx.x;
  const int tid = threadIdx.x, lane = tid & 31, wv = tid >> 5;
  const float* eb = e + (size_t)b * SEQ_LEN * HDIM;
  const float* qb = q + (size_t)b * HDIM;

  for (int h = tid; h < HDIM; h += 128) { sQ[h] = qb[h]; sV[h] = v[h]; }
  __syncthreads();

  for (int l = wv; l < SEQ_LEN; l += 4) {
    const float* el = eb + (size_t)l * HDIM;
    float p = 0.f;
    for (int h = lane; h < HDIM; h += 32)
      p += sV[h] * tanhf(sQ[h] + el[h]);
    for (int o = 16; o; o >>= 1) p += __shfl_xor(p, o, 32);
    if (lane == 0) sU[l] = mask[b * SEQ_LEN + l] ? -1e30f : 10.0f * tanhf(p);
  }
  __syncthreads();

  float u = sU[tid];
  float m = u;
  for (int o = 16; o; o >>= 1) m = fmaxf(m, __shfl_xor(m, o, 32));
  if (lane == 0) sR[wv] = m;
  __syncthreads();
  m = fmaxf(fmaxf(sR[0], sR[1]), fmaxf(sR[2], sR[3]));
  float ex = expf(u - m);
  float s = ex;
  for (int o = 16; o; o >>= 1) s += __shfl_xor(s, o, 32);
  __syncthreads();
  if (lane == 0) sR[wv] = s;
  __syncthreads();
  s = sR[0] + sR[1] + sR[2] + sR[3];
  probs_out[(size_t)b * SEQ_LEN + tid] = ex / s;

  // Gumbel-max categorical sample from logits u
  float z = u + gumbel_noise((unsigned)t, (unsigned)b, (unsigned)tid);
  int li = tid;
  for (int o = 16; o; o >>= 1) {
    float oz = __shfl_xor(z, o, 32);
    int   ol = __shfl_xor(li, o, 32);
    if (oz > z) { z = oz; li = ol; }
  }
  __syncthreads();
  if (lane == 0) { sR[wv] = z; sI[wv] = li; }
  __syncthreads();
  if (tid == 0) {
    float bz = sR[0]; int bidx = sI[0];
    for (int w = 1; w < 4; ++w) if (sR[w] > bz) { bz = sR[w]; bidx = sI[w]; }
    prev[b] = bidx;
    mask[b * SEQ_LEN + bidx] = 1u;
    sel_out[b] = (float)bidx;
  }
}

// ======================= misc =======================
__global__ void k_gather(const float* __restrict__ emb, const int* __restrict__ prev,
                         float* __restrict__ dec) {
  int b = blockIdx.x;
  int idx = prev[b];
  const float* src = emb + ((size_t)idx * BATCH + b) * HDIM;
  for (int h = threadIdx.x; h < HDIM; h += 128)
    dec[(size_t)b * HDIM + h] = src[h];
}

__global__ void k_init(const float* __restrict__ h0, const float* __restrict__ c0,
                       const float* __restrict__ dec0,
                       float* __restrict__ hx, float* __restrict__ cx, float* __restrict__ dec,
                       unsigned* __restrict__ mask, int* __restrict__ prev,
                       float* __restrict__ out_probs0, float* __restrict__ out_sel0) {
  int i = blockIdx.x * blockDim.x + threadIdx.x;   // 131072 threads
  hx[i] = h0[i]; cx[i] = c0[i]; dec[i] = dec0[i];
  if (i < BATCH * SEQ_LEN) {
    unsigned first = ((i & (SEQ_LEN - 1)) == 0) ? 1u : 0u;
    mask[i] = first;
    out_probs0[i] = first ? 1.0f : 0.0f;
  }
  if (i < BATCH) { prev[i] = 0; out_sel0[i] = 0.0f; }
}

__global__ void k_final(const float* __restrict__ hx, const float* __restrict__ cx,
                        float* __restrict__ oh, float* __restrict__ oc) {
  int i = blockIdx.x * blockDim.x + threadIdx.x;   // 131072 threads
  oh[i] = hx[i]; oc[i] = cx[i];
}

// ======================= host =======================
extern "C" void kernel_launch(void* const* d_in, const int* in_sizes, int n_in,
                              void* d_out, int out_size, void* d_ws, size_t ws_size,
                              hipStream_t stream) {
  (void)in_sizes; (void)n_in; (void)out_size; (void)ws_size;
  const float* dec0  = (const float*)d_in[0];
  const float* emb   = (const float*)d_in[1];
  const float* h0    = (const float*)d_in[2];
  const float* c0    = (const float*)d_in[3];
  const float* ctx   = (const float*)d_in[4];
  const float* Wi    = (const float*)d_in[5];
  const float* bi    = (const float*)d_in[6];
  const float* Wh    = (const float*)d_in[7];
  const float* bh    = (const float*)d_in[8];
  const float* gWq   = (const float*)d_in[9];
  const float* gbq   = (const float*)d_in[10];
  const float* gWref = (const float*)d_in[11];
  const float* gbref = (const float*)d_in[12];
  const float* gv    = (const float*)d_in[13];
  const float* pWq   = (const float*)d_in[14];
  const float* pbq   = (const float*)d_in[15];
  const float* pWref = (const float*)d_in[16];
  const float* pbref = (const float*)d_in[17];
  const float* pv    = (const float*)d_in[18];

  char* base = (char*)d_ws;
  size_t off = 0;
  auto take = [&](size_t bytes) -> char* {
    char* p = base + off;
    off = (off + bytes + 255) & ~(size_t)255;
    return p;
  };
  bf16_t*   wcat  = (bf16_t*)take((size_t)2048 * 1024 * 2);   // [Wi|Wh] bf16
  bf16_t*   gWq_b = (bf16_t*)take((size_t)512 * 512 * 2);
  bf16_t*   pWq_b = (bf16_t*)take((size_t)512 * 512 * 2);
  bf16_t*   gWr_b = (bf16_t*)take((size_t)512 * 512 * 2);
  bf16_t*   pWr_b = (bf16_t*)take((size_t)512 * 512 * 2);
  float*    bcat  = (float*)take(2048 * 4);
  float*    eg    = (float*)take((size_t)BATCH * SEQ_LEN * HDIM * 4);  // 64 MB (L2-resident)
  float*    ep    = (float*)take((size_t)BATCH * SEQ_LEN * HDIM * 4);  // 64 MB
  float*    gates = (float*)take((size_t)BATCH * 2048 * 4);
  float*    hx    = (float*)take((size_t)BATCH * HDIM * 4);
  float*    cx    = (float*)take((size_t)BATCH * HDIM * 4);
  float*    dec   = (float*)take((size_t)BATCH * HDIM * 4);
  float*    q     = (float*)take((size_t)BATCH * HDIM * 4);
  float*    gl    = (float*)take((size_t)BATCH * HDIM * 4);
  unsigned* mask  = (unsigned*)take((size_t)BATCH * SEQ_LEN * 4);
  int*      prev  = (int*)take((size_t)BATCH * 4);

  float* out_probs = (float*)d_out;                                    // (L,B,L)
  float* out_sel   = out_probs + (size_t)SEQ_LEN * BATCH * SEQ_LEN;    // (L,B)
  float* out_hy    = out_sel + (size_t)SEQ_LEN * BATCH;                // (B,H)
  float* out_cy    = out_hy + (size_t)BATCH * HDIM;                    // (B,H)

  // ---- weight prep (bf16) ----
  k_prep_wcat   <<<8192, 256, 0, stream>>>(Wi, Wh, wcat);
  k_cvt_bf16    <<<1024, 256, 0, stream>>>(gWq,   gWq_b, 512 * 512);
  k_cvt_bf16    <<<1024, 256, 0, stream>>>(pWq,   pWq_b, 512 * 512);
  k_cvt_bf16    <<<1024, 256, 0, stream>>>(gWref, gWr_b, 512 * 512);
  k_cvt_bf16    <<<1024, 256, 0, stream>>>(pWref, pWr_b, 512 * 512);
  k_prep_biascat<<<8, 256, 0, stream>>>(bi, bh, bcat);

  // ---- e_g / e_p precompute: M=L*B=32768, N=512, K=512, output permuted to [b][l][h] ----
  gemm_bf16_wmma<<<dim3(4, 512), 256, 0, stream>>>(ctx, ctx, 512, 512, 0,
                                                   gWr_b, gbref, eg, 512, 1);
  gemm_bf16_wmma<<<dim3(4, 512), 256, 0, stream>>>(ctx, ctx, 512, 512, 0,
                                                   pWr_b, pbref, ep, 512, 1);

  // ---- state init + step-0 outputs ----
  k_init<<<512, 256, 0, stream>>>(h0, c0, dec0, hx, cx, dec, mask, prev,
                                  out_probs, out_sel);

  // ---- sequential decode ----
  for (int t = 1; t < SEQ_LEN; ++t) {
    // gates = [dec|hx] @ [Wi;Wh]^T + (bi+bh) : M=256 N=2048 K=512+512
    gemm_bf16_wmma<<<dim3(16, 4), 256, 0, stream>>>(dec, hx, 512, 512, 512,
                                                    wcat, bcat, gates, 2048, 0);
    k_lstm_pw<<<512, 256, 0, stream>>>(gates, hx, cx);
    // glimpse query: q = hy @ gWq^T + gbq
    gemm_bf16_wmma<<<dim3(4, 4), 256, 0, stream>>>(hx, hx, 512, 512, 0,
                                                   gWq_b, gbq, q, 512, 0);
    k_glimpse<<<BATCH, 128, 0, stream>>>(eg, q, gv, mask, gl);
    // pointer query: q = gl @ pWq^T + pbq
    gemm_bf16_wmma<<<dim3(4, 4), 256, 0, stream>>>(gl, gl, 512, 512, 0,
                                                   pWq_b, pbq, q, 512, 0);
    k_pointer<<<BATCH, 128, 0, stream>>>(ep, q, pv, mask, prev,
                                         out_probs + (size_t)t * BATCH * SEQ_LEN,
                                         out_sel + (size_t)t * BATCH, t);
    k_gather<<<BATCH, 128, 0, stream>>>(emb, prev, dec);
  }

  k_final<<<512, 256, 0, stream>>>(hx, cx, out_hy, out_cy);
}
